// SwinTransBlock_61675730371236
// MI455X (gfx1250) — matrix operand order repaired and checked by reading, not compiled, not run on hardware
//
#include <hip/hip_runtime.h>
#include <math.h>
#include <stdint.h>

// ---------------------------------------------------------------------------
// CDNA5 (gfx1250) wave32 WMMA types
// ---------------------------------------------------------------------------
typedef __attribute__((ext_vector_type(16))) __bf16 v16bf;
typedef __attribute__((ext_vector_type(8)))  float  v8f;

#define DIMC   180
#define CP     192            // channel dim padded to 6*32 (also per-head 6*32)
#define HEADS  6
#define HD     30
#define HH     256
#define WWIDTH 256
#define BATCH  2
#define MTOK   (BATCH * HH * WWIDTH)   // 131072 tokens
#define NWIN_X 32
#define NWIN   (NWIN_X * NWIN_X)       // 1024 windows per image

// ---------------------------------------------------------------------------
// prep: per-token L2 norm; emit x and x/||x|| as bf16, row stride CP=192,
// padding columns zeroed. One wave per token.
// ---------------------------------------------------------------------------
__global__ __launch_bounds__(256) void prep_kernel(const float* __restrict__ x,
                                                   __bf16* __restrict__ xbf,
                                                   __bf16* __restrict__ nxbf) {
  int tok  = (blockIdx.x * blockDim.x + threadIdx.x) >> 5;
  int lane = threadIdx.x & 31;
  if (tok >= MTOK) return;
  const float* row = x + (size_t)tok * DIMC;
  float v[6];
  float ss = 0.f;
#pragma unroll
  for (int i = 0; i < 6; i++) {
    int c = lane + 32 * i;
    float t = (c < DIMC) ? row[c] : 0.f;
    v[i] = t;
    ss += t * t;
  }
#pragma unroll
  for (int off = 16; off > 0; off >>= 1) ss += __shfl_xor(ss, off, 32);
  float inv = 1.0f / fmaxf(sqrtf(ss), 1e-12f);
#pragma unroll
  for (int i = 0; i < 6; i++) {
    int c = lane + 32 * i;                 // covers all 192 padded columns
    size_t o = (size_t)tok * CP + c;
    xbf[o]  = (__bf16)v[i];
    nxbf[o] = (__bf16)(v[i] * inv);
  }
}

// ---------------------------------------------------------------------------
// wconv_pad: build padded, k-major, gain-folded bf16 weights.
//   B[kp][n], kp in [0,KP), n in [0,NT); zeros in all padding slots.
//   row map: headPad -> kp=(head*32+d) reads w in-dim head*30+d
//            plain   -> split at KP0 (A0 real dims Kin0, A1 real dims Kreal-Kin0)
//   col map: headPad -> n=(head*32+d) reads w out-dim head*30+d
// ---------------------------------------------------------------------------
__global__ void wconv_pad(const float* __restrict__ w, __bf16* __restrict__ bout,
                          int Nreal, int Kreal, int KP, int NT,
                          int Kin0, int KP0, int rowHeadPad, int colHeadPad,
                          float gain) {
  int idx = blockIdx.x * blockDim.x + threadIdx.x;
  if (idx >= KP * NT) return;
  int n = idx % NT, kp = idx / NT;
  int rk; bool rv;
  if (rowHeadPad) { int d = kp & 31; rk = (kp >> 5) * 30 + d; rv = (d < 30) && (rk < Kreal); }
  else if (kp < KP0) { rk = kp; rv = rk < Kin0; }
  else { rk = Kin0 + (kp - KP0); rv = rk < Kreal; }
  int rn; bool nv;
  if (colHeadPad) { int d = n & 31; rn = (n >> 5) * 30 + d; nv = (d < 30) && (rn < Nreal); }
  else { rn = n; nv = n < Nreal; }
  float val = (rv && nv) ? w[(size_t)rn * Kreal + rk] * gain : 0.f;
  bout[idx] = (__bf16)val;
}

__global__ void bias_pad(const float* __restrict__ b, float* __restrict__ out,
                         int Nreal, int NT, int colHeadPad) {
  int n = blockIdx.x * blockDim.x + threadIdx.x;
  if (n >= NT) return;
  int rn; bool nv;
  if (colHeadPad) { int d = n & 31; rn = (n >> 5) * 30 + d; nv = (d < 30) && (rn < Nreal); }
  else { rn = n; nv = n < Nreal; }
  out[n] = nv ? b[rn] : 0.f;
}

// ---------------------------------------------------------------------------
// GEMM: out = act(A @ B + bias), fully padded shapes (no guards in hot loop).
//   A: [M x KP] bf16 (optionally split in two stride-KP0/KP-KP0 arrays)
//   B: [KP x NT] bf16, zero-padded; bias: [NT] f32 zero-padded
//   tile: 128 rows x NT cols, 256 threads (8 waves, 16 rows each)
//   A staged via b128 loads, B via global_load_async_to_lds_b128 (ASYNCcnt),
//   next-tile global_prefetch, v_wmma_f32_16x16x32_bf16 accumulation,
//   epilogue staged through LDS and written as contiguous b128 stores.
// ---------------------------------------------------------------------------
template <int KTILES, int NTILES>
__global__ __launch_bounds__(256) void gemm_bf16(
    const __bf16* __restrict__ A0, const __bf16* __restrict__ A1, int KP0,
    const __bf16* __restrict__ B, const float* __restrict__ bias,
    float* __restrict__ outF, int NF,
    __bf16* __restrict__ outB, int act) {
  constexpr int KP   = KTILES * 32;
  constexpr int NT   = NTILES * 16;
  constexpr int BSTR = NT + 8;
  constexpr int CSTR = 200;                       // 192 + 8, bf16 elements
  __shared__ __align__(16) char smem[51200];      // A(10240)+B(<=25088) | C(51200)
  __bf16 (*As)[40]   = (__bf16 (*)[40])smem;      // 128 x 32 (stride 40)
  __bf16 (*Bs)[BSTR] = (__bf16 (*)[BSTR])(smem + 10240);
  __bf16* Cs = (__bf16*)smem;                     // epilogue reuse: 8*16*200*2

  int tid = threadIdx.x, wid = tid >> 5, lane = tid & 31;
  int mBase = blockIdx.x * 128;
  int lda1 = KP - KP0;

  v8f acc[NTILES];
#pragma unroll
  for (int t = 0; t < NTILES; t++) acc[t] = {};

  for (int kt = 0; kt < KTILES; kt++) {
    int k0 = kt * 32;
    const __bf16* Asrc; int lda, kk;
    if (k0 < KP0) { Asrc = A0; lda = KP0; kk = k0; }
    else          { Asrc = A1; lda = lda1; kk = k0 - KP0; }
    // ---- A tile: 128 x 32, two b128 loads + b128 LDS stores per thread ----
    {
      int r  = tid >> 1;
      int cb = (tid & 1) * 16;
      const uint4* src = (const uint4*)(Asrc + (size_t)(mBase + r) * lda + kk + cb);
      uint4* dst = (uint4*)&As[r][cb];
      dst[0] = src[0];
      dst[1] = src[1];
    }
    // ---- B tile: 32 x NT via async global->LDS b128 (ASYNCcnt path) ----
    {
      constexpr int U4PR = NT / 8;                // uint4 chunks per row
      for (int c = tid; c < 32 * U4PR; c += 256) {
        int row = c / U4PR, wi = c % U4PR;
        unsigned lds  = (unsigned)(uintptr_t)&Bs[row][wi * 8];
        unsigned voff = (unsigned)(((k0 + row) * NT + wi * 8) * sizeof(__bf16));
        asm volatile("global_load_async_to_lds_b128 %0, %1, %2"
                     :: "v"(lds), "v"(voff), "s"(B) : "memory");
      }
    }
    // ---- prefetch next A tile ----
    if (kt + 1 < KTILES) {
      int k0n = k0 + 32;
      const __bf16* An; int ldan, kkn;
      if (k0n < KP0) { An = A0; ldan = KP0; kkn = k0n; }
      else           { An = A1; ldan = lda1; kkn = k0n - KP0; }
      int r = tid >> 1, cb = (tid & 1) * 16;
      __builtin_prefetch(An + (size_t)(mBase + r) * ldan + kkn + cb, 0, 0);
    }
    asm volatile("s_wait_asynccnt 0" ::: "memory");
    __syncthreads();

    // ---- fragments + WMMA ----
    v16bf af;
    int rowl  = wid * 16 + (lane & 15);
    int khalf = (lane >> 4) * 8;
#pragma unroll
    for (int j = 0; j < 8; j++) {
      af[j]     = As[rowl][khalf + j];
      af[8 + j] = As[rowl][16 + khalf + j];
    }
#pragma unroll
    for (int t = 0; t < NTILES; t++) {
      v16bf bv;
      int col = t * 16 + (lane & 15);
      int kb  = (lane >> 4) * 8;
#pragma unroll
      for (int j = 0; j < 16; j++) bv[j] = Bs[kb + j + ((j >> 3) << 3)][col];
      acc[t] = __builtin_amdgcn_wmma_f32_16x16x32_bf16(
          false, af, false, bv, (short)0, acc[t], false, false);
    }
    __syncthreads();
  }

  // ---- epilogue: per-wave LDS staging, contiguous b128 global stores ----
  if (outB) {
    __bf16* Cw = Cs + wid * (16 * CSTR);
#pragma unroll
    for (int g = 0; g < NTILES / 12; g++) {
#pragma unroll
      for (int t = 0; t < 12; t++) {
        int tt  = g * 12 + t;
        int col = t * 16 + (lane & 15);
        float bcol = bias[tt * 16 + (lane & 15)];
#pragma unroll
        for (int i = 0; i < 8; i++) {
          int rl = (lane >> 4) * 8 + i;
          float y = acc[tt][i] + bcol;
          if (act) y = (y >= 0.f ? y : 0.2f * y) * 1.414213f;
          Cw[rl * CSTR + col] = (__bf16)y;
        }
      }
      __syncthreads();
#pragma unroll
      for (int c = 0; c < 12; c++) {              // 16 rows * 24 u4 / 32 lanes
        int idx = c * 32 + lane;
        int rl = idx / 24, wi = idx % 24;
        const uint4* s = (const uint4*)(Cw + rl * CSTR) + wi;
        size_t row = (size_t)(mBase + wid * 16 + rl);
        *((uint4*)(outB + row * NT + g * 192) + wi) = *s;
      }
      __syncthreads();
    }
  }
  if (outF) {                                      // final f32 output (unpadded)
#pragma unroll
    for (int t = 0; t < NTILES; t++) {
      int col = t * 16 + (lane & 15);
      if (col < NF) {
        float bcol = bias[t * 16 + (lane & 15)];
#pragma unroll
        for (int i = 0; i < 8; i++) {
          int row = mBase + wid * 16 + (lane >> 4) * 8 + i;
          float y = acc[t][i] + bcol;
          if (act) y = (y >= 0.f ? y : 0.2f * y) * 1.414213f;
          outF[(size_t)row * NF + col] = y;
        }
      }
    }
  }
}

// ---------------------------------------------------------------------------
// Window attention: one wave per (window, head); q/k/v are head-padded
// ([M x 192], head h occupies cols h*32..h*32+31, pads zero), so a head slice
// is one aligned 64B blob: 4 x b128 loads. QK^T (K=32), mask+softmax, attn@V
// (K=64) all via v_wmma_f32_16x16x32_bf16. Output written head-padded in
// original token order (window_reverse is free). Head 0 also emits mr.
// ---------------------------------------------------------------------------
__global__ __launch_bounds__(32) void attn_kernel(
    const __bf16* __restrict__ qg, const __bf16* __restrict__ kg,
    const __bf16* __restrict__ vg, const float* __restrict__ maskg,
    __bf16* __restrict__ outg, float* __restrict__ mrg) {
  __shared__ __align__(16) __bf16 qh[64][32];
  __shared__ __align__(16) __bf16 kh[64][32];
  __shared__ __align__(16) __bf16 vh[64][32];
  __shared__ float  attn[64][66];
  __shared__ __bf16 attnb[64][64];
  __shared__ float  amv[64];
  __shared__ int    rgn[64];

  int bid  = blockIdx.x;
  int head = bid % HEADS;
  int win  = bid / HEADS;
  int b    = win / NWIN;
  int wi   = win % NWIN;
  int wy   = wi / NWIN_X;
  int wx   = wi % NWIN_X;
  int lane = threadIdx.x;

  float rawsum = 0.f;
  for (int tt = 0; tt < 2; tt++) {
    int t = lane * 2 + tt;
    int h = wy * 8 + (t >> 3);
    int w = wx * 8 + (t & 7);
    size_t rowg = (size_t)b * (HH * WWIDTH) + (size_t)h * WWIDTH + w;
    size_t off  = rowg * CP + head * 32;
    const uint4* qp = (const uint4*)(qg + off);
    const uint4* kp = (const uint4*)(kg + off);
    const uint4* vp = (const uint4*)(vg + off);
    uint4* qd = (uint4*)&qh[t][0];
    uint4* kd = (uint4*)&kh[t][0];
    uint4* vd = (uint4*)&vh[t][0];
#pragma unroll
    for (int c = 0; c < 4; c++) { qd[c] = qp[c]; kd[c] = kp[c]; vd[c] = vp[c]; }
    float mval = maskg[rowg];
    rawsum += mval;
    amv[t] = (mval == 0.f) ? -100.f : 0.f;
    int rh = (h < HH - 8) ? 0 : ((h < HH - 4) ? 1 : 2);
    int rw = (w < WWIDTH - 8) ? 0 : ((w < WWIDTH - 4) ? 1 : 2);
    rgn[t] = rh * 3 + rw;
  }
  __syncthreads();

  const float scale = 0.18257418583505536f;  // 1/sqrt(30)

  // ---- QK^T : 4x4 tiles of 16x16, single K-step of 32 ----
  for (int mi = 0; mi < 4; mi++) {
    v16bf af;
    int rowl  = mi * 16 + (lane & 15);
    int khalf = (lane >> 4) * 8;
#pragma unroll
    for (int j = 0; j < 8; j++) {
      af[j]     = qh[rowl][khalf + j];
      af[8 + j] = qh[rowl][16 + khalf + j];
    }
    for (int ni = 0; ni < 4; ni++) {
      v16bf bv;
      int col = ni * 16 + (lane & 15);
      int kb  = (lane >> 4) * 8;
#pragma unroll
      for (int j = 0; j < 16; j++) bv[j] = kh[col][kb + j + ((j >> 3) << 3)];  // B[k][n]=K[n][k]
      v8f c = {};
      c = __builtin_amdgcn_wmma_f32_16x16x32_bf16(
          false, af, false, bv, (short)0, c, false, false);
#pragma unroll
      for (int i = 0; i < 8; i++) {
        int r = mi * 16 + (lane >> 4) * 8 + i;
        attn[r][ni * 16 + (lane & 15)] = c[i] * scale;
      }
    }
  }
  __syncthreads();

  // ---- masks + softmax (lane handles rows 2l, 2l+1) ----
  for (int tt = 0; tt < 2; tt++) {
    int r  = lane * 2 + tt;
    int rr = rgn[r];
    float mx = -1e30f;
    for (int j = 0; j < 64; j++) {
      float v = attn[r][j] + ((rgn[j] != rr) ? -100.f : 0.f) + amv[j];
      attn[r][j] = v;
      mx = fmaxf(mx, v);
    }
    float s = 0.f;
    for (int j = 0; j < 64; j++) {
      float e = __expf(attn[r][j] - mx);
      attn[r][j] = e;
      s += e;
    }
    float inv = 1.f / s;
    for (int j = 0; j < 64; j++) attnb[r][j] = (__bf16)(attn[r][j] * inv);
  }
  __syncthreads();

  // ---- attn @ V : 4x2 tiles, K = 64 (two 32-steps); pad cols give exact 0 ----
  for (int mi = 0; mi < 4; mi++) {
    for (int ni = 0; ni < 2; ni++) {
      v8f c = {};
      for (int ks = 0; ks < 2; ks++) {
        v16bf af;
        int rowl  = mi * 16 + (lane & 15);
        int kbase = ks * 32 + (lane >> 4) * 8;
#pragma unroll
        for (int j = 0; j < 8; j++) {
          af[j]     = attnb[rowl][kbase + j];
          af[8 + j] = attnb[rowl][kbase + 16 + j];
        }
        v16bf bv;
        int col = ni * 16 + (lane & 15);
        int kb  = ks * 32 + (lane >> 4) * 8;
#pragma unroll
        for (int j = 0; j < 16; j++) bv[j] = vh[kb + j + ((j >> 3) << 3)][col];
        c = __builtin_amdgcn_wmma_f32_16x16x32_bf16(
            false, af, false, bv, (short)0, c, false, false);
      }
      int d = ni * 16 + (lane & 15);
#pragma unroll
      for (int i = 0; i < 8; i++) {
        int t = mi * 16 + (lane >> 4) * 8 + i;
        int h = wy * 8 + (t >> 3);
        int w = wx * 8 + (t & 7);
        size_t rowg = (size_t)b * (HH * WWIDTH) + (size_t)h * WWIDTH + w;
        outg[rowg * CP + head * 32 + d] = (__bf16)c[i];
      }
    }
  }

  // ---- mr: clip(sum(mask over window), 0, 1) tiled to all 64 tokens ----
  if (head == 0) {
#pragma unroll
    for (int off = 16; off > 0; off >>= 1) rawsum += __shfl_xor(rawsum, off, 32);
    float mval = fminf(fmaxf(rawsum, 0.f), 1.f);
    for (int tt = 0; tt < 2; tt++) {
      int t = lane * 2 + tt;
      int h = wy * 8 + (t >> 3);
      int w = wx * 8 + (t & 7);
      mrg[(size_t)b * (HH * WWIDTH) + (size_t)h * WWIDTH + w] = mval;
    }
  }
}

// ---------------------------------------------------------------------------
// host orchestration
// ---------------------------------------------------------------------------
extern "C" void kernel_launch(void* const* d_in, const int* in_sizes, int n_in,
                              void* d_out, int out_size, void* d_ws, size_t ws_size,
                              hipStream_t stream) {
  (void)in_sizes; (void)n_in; (void)out_size; (void)ws_size;

  const float* x      = (const float*)d_in[0];
  const float* mask   = (const float*)d_in[1];
  const float* q_w    = (const float*)d_in[2];
  const float* q_b    = (const float*)d_in[3];
  const float* k_w    = (const float*)d_in[4];
  const float* k_b    = (const float*)d_in[5];
  const float* v_w    = (const float*)d_in[6];
  const float* v_b    = (const float*)d_in[7];
  const float* proj_w = (const float*)d_in[8];
  const float* proj_b = (const float*)d_in[9];
  const float* fuse_w = (const float*)d_in[10];
  const float* fuse_b = (const float*)d_in[11];
  const float* fc1_w  = (const float*)d_in[12];
  const float* fc1_b  = (const float*)d_in[13];
  const float* fc2_w  = (const float*)d_in[14];
  const float* fc2_b  = (const float*)d_in[15];

  float* yout  = (float*)d_out;
  float* mrout = yout + (size_t)MTOK * DIMC;

  const size_t U = (size_t)MTOK * CP;   // elements per padded [M x 192] buffer
  __bf16* ws = (__bf16*)d_ws;
  __bf16* xbf     = ws + 0 * U;   // shortcut, live through fuse
  __bf16* nxbf    = ws + 1 * U;   // normalized x -> reused as attn output
  __bf16* qbf     = ws + 2 * U;   // q (head-padded) -> reused as proj output
  __bf16* kbf     = ws + 3 * U;   // k (head-padded) -> reused as fuse output
  __bf16* vbf     = ws + 4 * U;   // v (head-padded) -> fc1 output (units 4..5)
  __bf16* attnout = nxbf;
  __bf16* projout = qbf;
  __bf16* fuseout = kbf;
  __bf16* fc1out  = vbf;          // [M x 384]
  __bf16* wts   = ws + 6 * U;
  __bf16* Bq    = wts;                   // [192 x 192] head-padded cols
  __bf16* Bk    = Bq    + 192 * 192;
  __bf16* Bv    = Bk    + 192 * 192;
  __bf16* Bproj = Bv    + 192 * 192;     // [192 x 192] head-padded rows
  __bf16* Bfuse = Bproj + 192 * 192;     // [384 x 192] split rows (x | xr)
  __bf16* Bfc1  = Bfuse + 384 * 192;     // [192 x 384]
  __bf16* Bfc2  = Bfc1  + 192 * 384;     // [384 x 192]
  float*  bq    = (float*)(Bfc2 + 384 * 192);
  float*  bk    = bq    + 192;
  float*  bv    = bk    + 192;
  float*  bproj = bv    + 192;
  float*  bfuse = bproj + 192;
  float*  bfc1  = bfuse + 192;           // [384]
  float*  bfc2  = bfc1  + 384;

  const float g180 = 1.0f / sqrtf(180.0f);
  const float g360 = 1.0f / sqrtf(360.0f);

  // 1) prep: norms + padded bf16 casts
  prep_kernel<<<MTOK / 8, 256, 0, stream>>>(x, xbf, nxbf);

  // 2) padded, gain-folded weights + biases
  const int WB = 256;
  wconv_pad<<<(192*192 + WB-1)/WB, WB, 0, stream>>>(q_w,    Bq,    180, 180, 192, 192, 180, 192, 0, 1, g180);
  wconv_pad<<<(192*192 + WB-1)/WB, WB, 0, stream>>>(k_w,    Bk,    180, 180, 192, 192, 180, 192, 0, 1, g180);
  wconv_pad<<<(192*192 + WB-1)/WB, WB, 0, stream>>>(v_w,    Bv,    180, 180, 192, 192, 180, 192, 0, 1, g180);
  wconv_pad<<<(192*192 + WB-1)/WB, WB, 0, stream>>>(proj_w, Bproj, 180, 180, 192, 192, 180, 192, 1, 0, g180);
  wconv_pad<<<(384*192 + WB-1)/WB, WB, 0, stream>>>(fuse_w, Bfuse, 180, 360, 384, 192, 180, 192, 0, 0, g360);
  wconv_pad<<<(192*384 + WB-1)/WB, WB, 0, stream>>>(fc1_w,  Bfc1,  360, 180, 192, 384, 180, 192, 0, 0, g180);
  wconv_pad<<<(384*192 + WB-1)/WB, WB, 0, stream>>>(fc2_w,  Bfc2,  180, 360, 384, 192, 360, 384, 0, 0, g360);
  bias_pad<<<1, 192, 0, stream>>>(q_b,    bq,    180, 192, 1);
  bias_pad<<<1, 192, 0, stream>>>(k_b,    bk,    180, 192, 1);
  bias_pad<<<1, 192, 0, stream>>>(v_b,    bv,    180, 192, 1);
  bias_pad<<<1, 192, 0, stream>>>(proj_b, bproj, 180, 192, 0);
  bias_pad<<<1, 192, 0, stream>>>(fuse_b, bfuse, 180, 192, 0);
  bias_pad<<<2, 256, 0, stream>>>(fc1_b,  bfc1,  360, 384, 0);
  bias_pad<<<1, 192, 0, stream>>>(fc2_b,  bfc2,  180, 192, 0);

  const int gblocks = MTOK / 128;

  // 3) QKV projections (WMMA), head-padded outputs
  gemm_bf16<6,12><<<gblocks, 256, 0, stream>>>(nxbf, nxbf, 192, Bq, bq,
                                               nullptr, 0, qbf, 0);
  gemm_bf16<6,12><<<gblocks, 256, 0, stream>>>(nxbf, nxbf, 192, Bk, bk,
                                               nullptr, 0, kbf, 0);
  gemm_bf16<6,12><<<gblocks, 256, 0, stream>>>(xbf, xbf, 192, Bv, bv,
                                               nullptr, 0, vbf, 0);

  // 4) window attention (WMMA), emits attnout (head-padded) + mr
  attn_kernel<<<BATCH * NWIN * HEADS, 32, 0, stream>>>(qbf, kbf, vbf, mask,
                                                       attnout, mrout);

  // 5) proj (rows head-padded in B)
  gemm_bf16<6,12><<<gblocks, 256, 0, stream>>>(attnout, attnout, 192, Bproj, bproj,
                                               nullptr, 0, projout, 0);
  // 6) fuse = lrelu(concat(shortcut, xr) @ Wf + b): split-K A (192 + 192)
  gemm_bf16<12,12><<<gblocks, 256, 0, stream>>>(xbf, projout, 192, Bfuse, bfuse,
                                                nullptr, 0, fuseout, 1);
  // 7) fc1 (N=360 -> NT=384)
  gemm_bf16<6,24><<<gblocks, 256, 0, stream>>>(fuseout, fuseout, 192, Bfc1, bfc1,
                                               nullptr, 0, fc1out, 1);
  // 8) fc2 -> final y (f32, unpadded, straight into d_out)
  gemm_bf16<12,12><<<gblocks, 256, 0, stream>>>(fc1out, fc1out, 384, Bfc2, bfc2,
                                                yout, DIMC, nullptr, 0);
}
